// FocalTransformerBlock_13795434955393
// MI455X (gfx1250) — matrix operand order repaired and checked by reading, not compiled, not run on hardware
//
#include <hip/hip_runtime.h>
#include <hip/hip_bf16.h>
#include <math.h>

// ---------------------------------------------------------------------------
// FocalTransformerBlock for MI455X (gfx1250, wave32, WMMA bf16 path)
// ---------------------------------------------------------------------------

typedef __bf16 bf16_t;
typedef __attribute__((ext_vector_type(16))) __bf16 v16bf;
typedef __attribute__((ext_vector_type(8)))  __bf16 v8bf;
typedef __attribute__((ext_vector_type(8)))  float  v8f;

#define B_      2
#define H_      256
#define W_      256
#define C_      128
#define WS_     8
#define NWH_    32
#define WA_     64
#define HEADS_  4
#define HD_     32
#define NTOK    (B_*H_*W_)        // 131072
#define NPOOL   (B_*NWH_*NWH_)    // 2048
#define NROLL   156
#define NALL    229               // 64 + 156 + 9
#define BIAS_STRIDE 240
#define SCALE_  0.17677669529663687f   // 32^-0.5

#define WMMA_BF16(a, b, c) \
  __builtin_amdgcn_wmma_f32_16x16x32_bf16(false, (a), false, (b), (short)0, (c), false, false)

__device__ inline v8f v8f_zero() {
  v8f z = {0.f, 0.f, 0.f, 0.f, 0.f, 0.f, 0.f, 0.f};
  return z;
}

__device__ inline v16bf load_afrag(const bf16_t* __restrict__ p, int hi) {
  // A-frag 16x32: per lane, K = hi*16..hi*16+15 (contiguous, matches B fragments).
  // shufflevector concat lets the two 4-VGPR load results sit adjacently (no moves).
  v8bf lo = *(const v8bf*)(p + hi * 16);
  v8bf hb = *(const v8bf*)(p + hi * 16 + 8);
  return __builtin_shufflevector(lo, hb, 0, 1, 2, 3, 4, 5, 6, 7,
                                 8, 9, 10, 11, 12, 13, 14, 15);
}

// ---------------------------------------------------------------------------
// Init: VALID_ROLLED index table + 64-element zero row for padded K fragments
// ---------------------------------------------------------------------------
__global__ void init_kernel(int* __restrict__ vtab, bf16_t* __restrict__ zrow) {
  if (threadIdx.x < 64) zrow[threadIdx.x] = (bf16_t)0.f;
  if (threadIdx.x == 0) {
    int n = 0;
    for (int q = 0; q < 4; q++)
      for (int r = 0; r < 8; r++)
        for (int c = 0; c < 8; c++) {
          bool zeroed;
          if (q == 0)      zeroed = (r < 5 && c < 5);
          else if (q == 1) zeroed = (r < 5 && c >= 3);
          else if (q == 2) zeroed = (r >= 3 && c < 5);
          else             zeroed = (r >= 3 && c >= 3);
          if (!zeroed) vtab[n++] = (q << 6) | (r << 3) | c;
        }
  }
}

// ---------------------------------------------------------------------------
// f32 -> bf16 weight conversion
// ---------------------------------------------------------------------------
__global__ void cvt_kernel(const float* __restrict__ in, bf16_t* __restrict__ out, int n) {
  int i = blockIdx.x * blockDim.x + threadIdx.x;
  if (i < n) out[i] = (bf16_t)in[i];
}

// ---------------------------------------------------------------------------
// LayerNorm over C=128: one wave32 per token, 4 channels per lane
// ---------------------------------------------------------------------------
__global__ __launch_bounds__(256) void ln_kernel(const float* __restrict__ x,
                                                 const float* __restrict__ g,
                                                 const float* __restrict__ b,
                                                 bf16_t* __restrict__ out, int ntok) {
  int wv = (blockIdx.x * blockDim.x + threadIdx.x) >> 5;
  int lane = threadIdx.x & 31;
  if (wv >= ntok) return;
  const float* row = x + (size_t)wv * C_;
  float v[4];
  float s = 0.f;
  #pragma unroll
  for (int i = 0; i < 4; i++) { v[i] = row[lane + 32 * i]; s += v[i]; }
  #pragma unroll
  for (int m = 16; m >= 1; m >>= 1) s += __shfl_xor(s, m, 32);
  float mu = s * (1.f / 128.f);
  float var = 0.f;
  #pragma unroll
  for (int i = 0; i < 4; i++) { float d = v[i] - mu; var += d * d; }
  #pragma unroll
  for (int m = 16; m >= 1; m >>= 1) var += __shfl_xor(var, m, 32);
  float rs = rsqrtf(var * (1.f / 128.f) + 1e-5f);
  bf16_t* orow = out + (size_t)wv * C_;
  #pragma unroll
  for (int i = 0; i < 4; i++) {
    int c = lane + 32 * i;
    orow[c] = (bf16_t)((v[i] - mu) * rs * g[c] + b[c]);
  }
}

// ---------------------------------------------------------------------------
// Window pooling: pooled[b,pi,pj,c] = sum_n xh[win(pi,pj,n),c]*pw[n] + pb
// ---------------------------------------------------------------------------
__global__ __launch_bounds__(256) void pool_kernel(const bf16_t* __restrict__ xh,
                                                   const float* __restrict__ pw,
                                                   const float* __restrict__ pb,
                                                   bf16_t* __restrict__ pooled) {
  int idx = blockIdx.x * blockDim.x + threadIdx.x;
  if (idx >= NPOOL * C_) return;
  int c = idx & (C_ - 1);
  int p = idx / C_;
  int b = p >> 10;
  int pi = (p >> 5) & 31;
  int pj = p & 31;
  float acc = 0.f;
  #pragma unroll 8
  for (int n = 0; n < 64; n++) {
    int r = n >> 3, cc = n & 7;
    size_t row = ((size_t)b * H_ + pi * 8 + r) * W_ + pj * 8 + cc;
    acc += (float)xh[row * C_ + c] * pw[n];
  }
  pooled[(size_t)p * C_ + c] = (bf16_t)(acc + pb[0]);
}

// ---------------------------------------------------------------------------
// Relative position bias table: bias_all[h][q][t], t < 229, stride 240
// ---------------------------------------------------------------------------
__global__ void bias_kernel(const float* __restrict__ rpb_table,
                            const float* __restrict__ rpb_nb,
                            const float* __restrict__ rpb_w0,
                            float* __restrict__ bias_all) {
  for (int idx = blockIdx.x * blockDim.x + threadIdx.x; idx < HEADS_ * WA_ * NALL;
       idx += gridDim.x * blockDim.x) {
    int t = idx % NALL;
    int q = (idx / NALL) % WA_;
    int h = idx / (NALL * WA_);
    int qr = q >> 3, qc = q & 7;
    float v;
    if (t < 64) {
      int kr = t >> 3, kc = t & 7;
      v = rpb_table[((qr - kr + 7) * 15 + (qc - kc + 7)) * HEADS_ + h];
    } else if (t < 220) {
      v = rpb_nb[((size_t)(h * WA_ + q)) * NROLL + (t - 64)];
    } else {
      int k = t - 220;
      int kr = k / 3, kc = k % 3;
      v = rpb_w0[h * 100 + (qr - kr + 2) * 10 + (qc - kc + 2)];
    }
    bias_all[(size_t)(h * WA_ + q) * BIAS_STRIDE + t] = v;
  }
}

// ---------------------------------------------------------------------------
// GEMM: out[m,n] = sum_k A[m,k]*W[n,k] + bias[n]  (+epilogue)
// Block 256 = 8 waves; block tile 128(M) x 64(N); wave tile 32x32; K step 32.
// EPI 0: bf16 out   EPI 1: f32 out + resid   EPI 2: bf16 exact-GELU out
// ---------------------------------------------------------------------------
template <int EPI>
__global__ __launch_bounds__(256) void gemm_kernel(const bf16_t* __restrict__ A,
                                                   const bf16_t* __restrict__ Wt,
                                                   const float* __restrict__ bias,
                                                   const float* __restrict__ resid,
                                                   void* __restrict__ outp,
                                                   int M, int N, int K) {
  int lane = threadIdx.x & 31;
  int wave = threadIdx.x >> 5;
  int hi = lane >> 4, l15 = lane & 15;
  int wm = wave & 3, wn = wave >> 2;
  int m0 = blockIdx.y * 128 + wm * 32;
  int n0 = blockIdx.x * 64 + wn * 32;
  (void)M;

  v8f acc[2][2];
  #pragma unroll
  for (int im = 0; im < 2; im++)
    #pragma unroll
    for (int in = 0; in < 2; in++) acc[im][in] = v8f_zero();

  for (int k0 = 0; k0 < K; k0 += 32) {
    v16bf a[2], bW[2];
    #pragma unroll
    for (int im = 0; im < 2; im++)
      a[im] = load_afrag(A + (size_t)(m0 + im * 16 + l15) * K + k0, hi);
    #pragma unroll
    for (int in = 0; in < 2; in++)
      bW[in] = *(const v16bf*)(Wt + (size_t)(n0 + in * 16 + l15) * K + k0 + hi * 16);
    #pragma unroll
    for (int im = 0; im < 2; im++)
      #pragma unroll
      for (int in = 0; in < 2; in++) acc[im][in] = WMMA_BF16(a[im], bW[in], acc[im][in]);
  }

  #pragma unroll
  for (int im = 0; im < 2; im++)
    #pragma unroll
    for (int in = 0; in < 2; in++) {
      int col = n0 + in * 16 + l15;
      float bb = bias[col];
      #pragma unroll
      for (int i = 0; i < 8; i++) {
        int row = m0 + im * 16 + i + 8 * hi;
        size_t o = (size_t)row * N + col;
        float v = acc[im][in][i] + bb;
        if (EPI == 0) {
          ((bf16_t*)outp)[o] = (bf16_t)v;
        } else if (EPI == 1) {
          ((float*)outp)[o] = v + resid[o];
        } else {
          float g = 0.5f * v * (1.f + erff(v * 0.70710678118654752f));
          ((bf16_t*)outp)[o] = (bf16_t)g;
        }
      }
    }
}

// ---------------------------------------------------------------------------
// Attention helpers: pointer to 32-element head slice of key/value token t
// ---------------------------------------------------------------------------
__device__ inline const bf16_t* kv_ptr(const bf16_t* __restrict__ qkv,
                                       const bf16_t* __restrict__ qkvp,
                                       const int* __restrict__ vtab,
                                       int b, int wi, int wj, int t, int coff) {
  if (t < 64) {
    int r = t >> 3, c = t & 7;
    size_t row = ((size_t)b << 16) + (size_t)(wi * 8 + r) * 256 + (wj * 8 + c);
    return qkv + row * 384 + coff;
  } else if (t < 220) {
    int pk = vtab[t - 64];
    int q = pk >> 6, r = (pk >> 3) & 7, c = pk & 7;
    int sh = (q < 2) ? -3 : 3;
    int sw = (q & 1) ? 3 : -3;
    int src_h = (wi * 8 + r - sh + 256) & 255;
    int src_w = (wj * 8 + c - sw + 256) & 255;
    size_t row = ((size_t)b << 16) + (size_t)src_h * 256 + src_w;
    return qkv + row * 384 + coff;
  } else if (t < 229) {
    int k = t - 220;
    int pi = wi + k / 3 - 1, pj = wj + (k % 3) - 1;
    if (pi >= 0 && pi < 32 && pj >= 0 && pj < 32)
      return qkvp + ((size_t)b * 1024 + pi * 32 + pj) * 384 + coff;
    return nullptr;
  }
  return nullptr;
}

// ---------------------------------------------------------------------------
// Attention: one block per (window, head). 128 threads = 4 waves,
// wave wv owns query rows [16*wv, 16*wv+16).
//   S = Q*K^T (15 WMMA N-tiles, K=32) -> scale+bias+mask -> register softmax
//   (16-lane shfl reductions, D-layout) -> P to LDS -> O = P*V (16 WMMAs).
// ---------------------------------------------------------------------------
__global__ __launch_bounds__(128) void attn_kernel(const bf16_t* __restrict__ qkv,
                                                   const bf16_t* __restrict__ qkvp,
                                                   const float* __restrict__ bias_all,
                                                   const int* __restrict__ vtab,
                                                   const bf16_t* __restrict__ zrow,
                                                   bf16_t* __restrict__ out) {
  __shared__ bf16_t vT[32][256];        // V^T gather, zero padded in t
  __shared__ bf16_t Pbuf[4][16][256];   // per-wave softmax(P) strip

  int bid = blockIdx.x;
  int h = bid & 3;
  int w = bid >> 2;
  int b = w >> 10;
  int wi = (w >> 5) & 31;
  int wj = w & 31;
  int tid = threadIdx.x;
  int lane = tid & 31;
  int wv = tid >> 5;
  int hi = lane >> 4, l15 = lane & 15;
  int koff = C_ + h * HD_;       // key channel offset within qkv row
  int voff = 2 * C_ + h * HD_;   // value channel offset

  // --- gather V^T into LDS (t in [0,256), pads are zero) ---
  for (int e = tid; e < 32 * 256; e += 128) {
    int t = e >> 5;
    int d = e & 31;
    bf16_t bv = (bf16_t)0.f;
    const bf16_t* p = kv_ptr(qkv, qkvp, vtab, b, wi, wj, t, voff);
    if (p) bv = p[d];
    vT[d][t] = bv;
  }
  __syncthreads();

  // --- Q A-fragment straight from global ---
  int qtok = 16 * wv + l15;
  int qr = qtok >> 3, qc = qtok & 7;
  size_t qrow = ((size_t)b << 16) + (size_t)(wi * 8 + qr) * 256 + (wj * 8 + qc);
  v16bf aq = load_afrag(qkv + qrow * 384 + h * HD_, hi);

  // --- S = Q * K^T ---
  v8f sv[15];
  #pragma unroll
  for (int jt = 0; jt < 15; jt++) {
    int t = 16 * jt + l15;
    const bf16_t* kp = kv_ptr(qkv, qkvp, vtab, b, wi, wj, t, koff);
    if (!kp) kp = zrow;
    v16bf bk = *(const v16bf*)(kp + hi * 16);
    sv[jt] = WMMA_BF16(aq, bk, v8f_zero());
  }

  // --- scale + bias + pool mask / padding mask ---
  #pragma unroll
  for (int jt = 0; jt < 15; jt++) {
    int t = 16 * jt + l15;
    float mask = 0.f;
    bool valid = (t < NALL);
    if (t >= 220 && t < NALL) {
      int k = t - 220;
      int pi = wi + k / 3 - 1, pj = wj + (k % 3) - 1;
      mask = (pi >= 0 && pi < 32 && pj >= 0 && pj < 32) ? 0.f : -100.f;
    }
    #pragma unroll
    for (int i = 0; i < 8; i++) {
      int q = 16 * wv + i + 8 * hi;
      float v;
      if (valid)
        v = sv[jt][i] * SCALE_ + bias_all[(size_t)(h * WA_ + q) * BIAS_STRIDE + t] + mask;
      else
        v = -1e30f;
      sv[jt][i] = v;
    }
  }

  // --- softmax over t: rows live in 16-lane half-groups of D layout ---
  #pragma unroll
  for (int i = 0; i < 8; i++) {
    float m = -1e30f;
    #pragma unroll
    for (int jt = 0; jt < 15; jt++) m = fmaxf(m, sv[jt][i]);
    #pragma unroll
    for (int d = 1; d < 16; d <<= 1) m = fmaxf(m, __shfl_xor(m, d, 32));
    float s = 0.f;
    #pragma unroll
    for (int jt = 0; jt < 15; jt++) {
      float p = expf(sv[jt][i] - m);
      sv[jt][i] = p;
      s += p;
    }
    #pragma unroll
    for (int d = 1; d < 16; d <<= 1) s += __shfl_xor(s, d, 32);
    float inv = 1.f / s;
    #pragma unroll
    for (int jt = 0; jt < 15; jt++)
      Pbuf[wv][i + 8 * hi][16 * jt + l15] = (bf16_t)(sv[jt][i] * inv);
    // zero the padded K columns [240, 256)
    Pbuf[wv][i + 8 * hi][240 + l15] = (bf16_t)0.f;
  }
  __syncthreads();

  // --- O = P * V (K = 256 padded, 8 steps of 32) ---
  v8f o0 = v8f_zero(), o1 = v8f_zero();
  #pragma unroll
  for (int ks = 0; ks < 8; ks++) {
    v16bf pa = load_afrag(&Pbuf[wv][l15][ks * 32], hi);
    v16bf vb0 = *(const v16bf*)(&vT[l15][ks * 32 + hi * 16]);
    v16bf vb1 = *(const v16bf*)(&vT[16 + l15][ks * 32 + hi * 16]);
    o0 = WMMA_BF16(pa, vb0, o0);
    o1 = WMMA_BF16(pa, vb1, o1);
  }

  // --- scatter O back to pixel-ordered (b,h,w, head*32+d) bf16 buffer ---
  #pragma unroll
  for (int i = 0; i < 8; i++) {
    int q = 16 * wv + i + 8 * hi;
    int r = q >> 3, c = q & 7;
    size_t grow = ((size_t)b << 16) + (size_t)(wi * 8 + r) * 256 + (wj * 8 + c);
    out[grow * 128 + h * HD_ + l15] = (bf16_t)o0[i];
    out[grow * 128 + h * HD_ + 16 + l15] = (bf16_t)o1[i];
  }
}

// ---------------------------------------------------------------------------
// Host launch
// ---------------------------------------------------------------------------
extern "C" void kernel_launch(void* const* d_in, const int* in_sizes, int n_in,
                              void* d_out, int out_size, void* d_ws, size_t ws_size,
                              hipStream_t stream) {
  (void)in_sizes; (void)n_in; (void)out_size; (void)ws_size;
  const float* x      = (const float*)d_in[0];
  const float* qkv_w  = (const float*)d_in[1];
  const float* qkv_b  = (const float*)d_in[2];
  const float* proj_w = (const float*)d_in[3];
  const float* proj_b = (const float*)d_in[4];
  const float* n1g    = (const float*)d_in[5];
  const float* n1b    = (const float*)d_in[6];
  const float* n2g    = (const float*)d_in[7];
  const float* n2b    = (const float*)d_in[8];
  const float* rpb_t  = (const float*)d_in[9];
  const float* rpb_nb = (const float*)d_in[10];
  const float* rpb_w0 = (const float*)d_in[11];
  const float* pool_w = (const float*)d_in[12];
  const float* pool_b = (const float*)d_in[13];
  const float* fc1_w  = (const float*)d_in[14];
  const float* fc1_b  = (const float*)d_in[15];
  const float* fc2_w  = (const float*)d_in[16];
  const float* fc2_b  = (const float*)d_in[17];

  char* ws = (char*)d_ws;
  size_t off = 0;
  auto alloc = [&](size_t bytes) {
    size_t o = off;
    off += (bytes + 255) & ~(size_t)255;
    return o;
  };

  // xh (also reused for LN2 output "h")
  bf16_t* xh = (bf16_t*)(ws + alloc((size_t)NTOK * C_ * 2));
  // grouped region: qkv | pooled | qkv_pooled | attn_out  (later aliased by GELU out)
  size_t grp_bytes = (size_t)NTOK * 384 * 2 + (size_t)NPOOL * C_ * 2 +
                     (size_t)NPOOL * 384 * 2 + (size_t)NTOK * C_ * 2;
  char* grp = ws + alloc(grp_bytes);
  bf16_t* qkv    = (bf16_t*)grp;
  bf16_t* pooled = (bf16_t*)(grp + (size_t)NTOK * 384 * 2);
  bf16_t* qkvp   = (bf16_t*)(grp + (size_t)NTOK * 384 * 2 + (size_t)NPOOL * C_ * 2);
  bf16_t* attn_o = (bf16_t*)(grp + (size_t)NTOK * 384 * 2 + (size_t)NPOOL * C_ * 2 +
                             (size_t)NPOOL * 384 * 2);
  bf16_t* gelu_o = (bf16_t*)grp;  // alias: NTOK*512*2 <= grp_bytes, qkv dead by then
  float*  x1     = (float*)(ws + alloc((size_t)NTOK * C_ * 4));
  bf16_t* wqkv   = (bf16_t*)(ws + alloc((size_t)384 * 128 * 2));
  bf16_t* wproj  = (bf16_t*)(ws + alloc((size_t)128 * 128 * 2));
  bf16_t* wfc1   = (bf16_t*)(ws + alloc((size_t)512 * 128 * 2));
  bf16_t* wfc2   = (bf16_t*)(ws + alloc((size_t)128 * 512 * 2));
  float*  biasA  = (float*)(ws + alloc((size_t)HEADS_ * WA_ * BIAS_STRIDE * 4));
  int*    vtab   = (int*)(ws + alloc(NROLL * 4));
  bf16_t* zrow   = (bf16_t*)(ws + alloc(64 * 2));

  // --- prep ---
  init_kernel<<<1, 64, 0, stream>>>(vtab, zrow);
  cvt_kernel<<<(384 * 128 + 255) / 256, 256, 0, stream>>>(qkv_w, wqkv, 384 * 128);
  cvt_kernel<<<(128 * 128 + 255) / 256, 256, 0, stream>>>(proj_w, wproj, 128 * 128);
  cvt_kernel<<<(512 * 128 + 255) / 256, 256, 0, stream>>>(fc1_w, wfc1, 512 * 128);
  cvt_kernel<<<(128 * 512 + 255) / 256, 256, 0, stream>>>(fc2_w, wfc2, 128 * 512);
  bias_kernel<<<64, 256, 0, stream>>>(rpb_t, rpb_nb, rpb_w0, biasA);

  // --- LN1, pooling ---
  ln_kernel<<<NTOK / 8, 256, 0, stream>>>(x, n1g, n1b, xh, NTOK);
  pool_kernel<<<(NPOOL * C_) / 256, 256, 0, stream>>>(xh, pool_w, pool_b, pooled);

  // --- QKV projections (main map + pooled tokens) ---
  gemm_kernel<0><<<dim3(384 / 64, NTOK / 128), 256, 0, stream>>>(
      xh, wqkv, qkv_b, nullptr, qkv, NTOK, 384, 128);
  gemm_kernel<0><<<dim3(384 / 64, NPOOL / 128), 256, 0, stream>>>(
      pooled, wqkv, qkv_b, nullptr, qkvp, NPOOL, 384, 128);

  // --- focal windowed attention ---
  attn_kernel<<<B_ * NWH_ * NWH_ * HEADS_, 128, 0, stream>>>(
      qkv, qkvp, biasA, vtab, zrow, attn_o);

  // --- proj + residual -> x1 (f32) ---
  gemm_kernel<1><<<dim3(128 / 64, NTOK / 128), 256, 0, stream>>>(
      attn_o, wproj, proj_b, x, x1, NTOK, 128, 128);

  // --- LN2 -> h (bf16, reuses xh) ---
  ln_kernel<<<NTOK / 8, 256, 0, stream>>>(x1, n2g, n2b, xh, NTOK);

  // --- MLP: fc1 + exact GELU, then fc2 + bias + residual -> d_out (f32) ---
  gemm_kernel<2><<<dim3(512 / 64, NTOK / 128), 256, 0, stream>>>(
      xh, wfc1, fc1_b, nullptr, gelu_o, NTOK, 512, 128);
  gemm_kernel<1><<<dim3(128 / 64, NTOK / 128), 256, 0, stream>>>(
      gelu_o, wfc2, fc2_b, x1, (float*)d_out, NTOK, 128, 512);
}
